// AttentionStreamBlock_16544214024451
// MI455X (gfx1250) — matrix operand order repaired and checked by reading, not compiled, not run on hardware
//
#include <hip/hip_runtime.h>
#include <hip/hip_bf16.h>

// ---------------------------------------------------------------------------
// Types for CDNA5 WMMA (wave32, 16x16x32 bf16 -> f32)
// ---------------------------------------------------------------------------
typedef __attribute__((ext_vector_type(16))) __bf16 v16bf;
typedef __attribute__((ext_vector_type(8)))  float  v8f;

union frag16 { v16bf v; unsigned int u[8]; };

__device__ __forceinline__ unsigned short f32_to_bf16(float f) {
    unsigned int u = __float_as_uint(f);
    u += 0x7FFFu + ((u >> 16) & 1u);          // round-to-nearest-even
    return (unsigned short)(u >> 16);
}
__device__ __forceinline__ unsigned int pack_bf16(float x, float y) {
    return (unsigned int)f32_to_bf16(x) | ((unsigned int)f32_to_bf16(y) << 16);
}

// ---------------------------------------------------------------------------
// f32 -> bf16 conversion pass (vectorized 4-wide). n must be a multiple of 4.
// ---------------------------------------------------------------------------
__global__ void cvt_bf16_kernel(const float4* __restrict__ in, uint2* __restrict__ out,
                                int n4) {
    int i = blockIdx.x * blockDim.x + threadIdx.x;
    if (i < n4) {
        float4 f = in[i];
        out[i] = make_uint2(pack_bf16(f.x, f.y), pack_bf16(f.z, f.w));
    }
}

// ---------------------------------------------------------------------------
// Elementwise: shifted = mu*x + (1-mu)*x_prev
// ---------------------------------------------------------------------------
__global__ void shift_kernel(const float* __restrict__ x, const float* __restrict__ xp,
                             const float* __restrict__ mu, float* __restrict__ out, int n) {
    int i = blockIdx.x * blockDim.x + threadIdx.x;
    if (i < n) {
        float m = mu[0];
        out[i] = m * x[i] + (1.f - m) * xp[i];
    }
}

// ---------------------------------------------------------------------------
// Elementwise: out = sigmoid(r) * v
// ---------------------------------------------------------------------------
__global__ void sigmul_kernel(const float* __restrict__ r, const float* __restrict__ v,
                              float* __restrict__ out, int n) {
    int i = blockIdx.x * blockDim.x + threadIdx.x;
    if (i < n) {
        float s = 1.f / (1.f + __expf(-r[i]));
        out[i] = s * v[i];
    }
}

// ---------------------------------------------------------------------------
// LayerNorm over D=512, one block (256 threads) per row; optional residual.
// Reads into registers first, so in-place (out == in) is safe.
// ---------------------------------------------------------------------------
__global__ void __launch_bounds__(256)
ln_kernel(const float* __restrict__ in, const float* __restrict__ res,
          const float* __restrict__ g, const float* __restrict__ beta,
          float* __restrict__ out) {
    const int row = blockIdx.x;
    const int t = threadIdx.x;
    const float* p = in + (size_t)row * 512;
    float v0 = p[t], v1 = p[t + 256];
    if (res) {
        const float* q = res + (size_t)row * 512;
        v0 += q[t]; v1 += q[t + 256];
    }
    __shared__ float s0[256], s1[256];
    s0[t] = v0 + v1;
    s1[t] = v0 * v0 + v1 * v1;
    __syncthreads();
    for (int s = 128; s > 0; s >>= 1) {
        if (t < s) { s0[t] += s0[t + s]; s1[t] += s1[t + s]; }
        __syncthreads();
    }
    float mean = s0[0] * (1.f / 512.f);
    float var  = s1[0] * (1.f / 512.f) - mean * mean;
    float inv  = rsqrtf(var + 1e-5f);
    float* o = out + (size_t)row * 512;
    o[t]       = (v0 - mean) * inv * g[t]       + beta[t];
    o[t + 256] = (v1 - mean) * inv * g[t + 256] + beta[t + 256];
}

// ---------------------------------------------------------------------------
// Sequential decayed-scan (SequenceMergingSeq). One block per batch,
// 512 threads = channels; per-step LDS reductions for mean(decay), sum(e).
// ---------------------------------------------------------------------------
__global__ void __launch_bounds__(512)
scan_kernel(const float* __restrict__ x1, const float* __restrict__ tm,
            float* __restrict__ outs) {
    const int b = blockIdx.x;
    const int d = threadIdx.x;
    const float tmd = tm[d];
    float bacc = 0.f, a = 0.f;
    __shared__ float sd[512], se[512];
    const float* xp = x1 + (size_t)b * 2048 * 512;
    float* op = outs + (size_t)b * 2048 * 512;
    for (int t = 0; t < 2048; ++t) {
        float xt = xp[(size_t)t * 512 + d];
        float decay = 1.f / (1.f + __expf(-xt * tmd));
        float e = __expf(xt);
        sd[d] = decay; se[d] = e;
        __syncthreads();
        for (int s = 256; s > 0; s >>= 1) {
            if (d < s) { sd[d] += sd[d + s]; se[d] += se[d + s]; }
            __syncthreads();
        }
        float dmean = sd[0] * (1.f / 512.f);
        float esum  = se[0];
        __syncthreads();                     // protect sd/se for next step
        a = dmean * a + esum;
        bacc = decay * bacc + e * xt;
        op[(size_t)t * 512 + d] = bacc / (a + 1e-8f);
    }
}

// ---------------------------------------------------------------------------
// WMMA GEMM, all-bf16 operands: C[M,N] = act(A[M,K] @ W[N,K]^T + bias[N]).
// A, W are bf16 (pre-converted); C is f32.
// Block: 256 threads = 8 waves, 128x64 output tile, K-step 64.
// Double-buffered bf16 LDS tiles filled by GLOBAL_LOAD_ASYNC_TO_LDS_B128;
// ASYNCcnt split-phase: issue tile k+1's 6 copies, s_wait_asynccnt 6 for tile k.
// Inner loop is pure ds_load_b128 fragments + v_wmma (no conversion VALU);
// each wave: 8 WMMAs per barrier round, A-fragment reused across 4 B tiles.
// act: 0 = none, 1 = relu.
// ---------------------------------------------------------------------------
__global__ void __launch_bounds__(256)
gemm_wmma_kernel(const unsigned short* __restrict__ A, const unsigned short* __restrict__ W,
                 const float* __restrict__ bias, float* __restrict__ C,
                 int M, int N, int K, int act) {
    __shared__ unsigned short As[2][128 * 64];  // 2 x 16KB, row-major [128][64]
    __shared__ unsigned short Ws[2][64 * 64];   // 2 x  8KB, row-major [64][64]
    const int tid  = threadIdx.x;
    const int lane = tid & 31;
    const int wave = tid >> 5;           // 0..7 = M sub-tile of 16
    const int half = (lane >> 4) & 1;
    const int l16  = lane & 15;
    const int m0 = blockIdx.y * 128;
    const int n0 = blockIdx.x * 64;

    // per-thread async-copy coordinates (A: 32 bf16 = 4xB128, W: 16 = 2xB128)
    const int ea = tid * 32, ra = ea >> 6, ca = ea & 63;
    const int ew = tid * 16, rw = ew >> 6, cw = ew & 63;

    // Issue one K-tile's copies into buffer `buf`. INST_OFFSET is added to both
    // the global and LDS addresses, so one base pair + offset covers each chunk.
    auto issue_tile = [&](int buf, int k0) {
        unsigned int la = (unsigned int)(unsigned long long)&As[buf][ea];
        unsigned long long ga =
            (unsigned long long)(const void*)(A + (size_t)(m0 + ra) * K + k0 + ca);
        asm volatile("global_load_async_to_lds_b128 %0, %1, off offset:0"  :: "v"(la), "v"(ga) : "memory");
        asm volatile("global_load_async_to_lds_b128 %0, %1, off offset:16" :: "v"(la), "v"(ga) : "memory");
        asm volatile("global_load_async_to_lds_b128 %0, %1, off offset:32" :: "v"(la), "v"(ga) : "memory");
        asm volatile("global_load_async_to_lds_b128 %0, %1, off offset:48" :: "v"(la), "v"(ga) : "memory");
        unsigned int lw = (unsigned int)(unsigned long long)&Ws[buf][ew];
        unsigned long long gw =
            (unsigned long long)(const void*)(W + (size_t)(n0 + rw) * K + k0 + cw);
        asm volatile("global_load_async_to_lds_b128 %0, %1, off offset:0"  :: "v"(lw), "v"(gw) : "memory");
        asm volatile("global_load_async_to_lds_b128 %0, %1, off offset:16" :: "v"(lw), "v"(gw) : "memory");
    };

    v8f acc[4] = {};
    issue_tile(0, 0);                                  // prime the pipeline
    for (int k0 = 0; k0 < K; k0 += 64) {
        const int cur = (k0 >> 6) & 1;
        if (k0 + 64 < K) {
            issue_tile(cur ^ 1, k0 + 64);              // 6 copies now in flight
            asm volatile("s_wait_asynccnt 6" ::: "memory");   // tile k landed
        } else {
            asm volatile("s_wait_asynccnt 0" ::: "memory");
        }
        __syncthreads();

        #pragma unroll
        for (int kk = 0; kk < 64; kk += 32) {
            frag16 a;
            #pragma unroll
            for (int j = 0; j < 8; ++j) {
                // A 16x32 layout: elem pair j -> K = 2*(j%4) + half*8 + (j/4)*16
                int ka = 2 * (j & 3) + half * 8 + ((j >> 2) << 4) + kk;
                a.u[j] = *(const unsigned int*)&As[cur][(wave * 16 + l16) * 64 + ka];
            }
            #pragma unroll
            for (int t = 0; t < 4; ++t) {
                frag16 b;
                #pragma unroll
                for (int j = 0; j < 8; ++j) {
                    // B 32x16 layout: elem pair j -> K = 2*(j%4) + half*8 + (j/4)*16, col = lane%16
                    int kb = 2 * (j & 3) + half * 8 + ((j >> 2) << 4) + kk;
                    b.u[j] = *(const unsigned int*)&Ws[cur][(t * 16 + l16) * 64 + kb];
                }
                acc[t] = __builtin_amdgcn_wmma_f32_16x16x32_bf16(false, a.v, false, b.v,
                                                                 (short)0, acc[t], false, false);
            }
        }
        __syncthreads();
    }
    // Epilogue. C layout: vgpr r -> row = r + 8*half, col = lane%16.
    #pragma unroll
    for (int t = 0; t < 4; ++t) {
        const int col = n0 + t * 16 + l16;
        const float bn = bias ? bias[col] : 0.f;
        #pragma unroll
        for (int r = 0; r < 8; ++r) {
            int row = m0 + wave * 16 + half * 8 + r;
            float v = acc[t][r] + bn;
            if (act == 1) v = fmaxf(v, 0.f);
            C[(size_t)row * N + col] = v;
        }
    }
}

// ---------------------------------------------------------------------------
// Flash attention. qkv: [B,S,3*512] f32 (q|k|v each 512, head h at h*64).
// Grid (S/128, H, B); block 256 = 8 waves; wave owns 16 queries.
// QK^T and AV both via WMMA; online softmax on C-fragment rows via shfl_xor.
// ---------------------------------------------------------------------------
__global__ void __launch_bounds__(256)
flash_attn_kernel(const float* __restrict__ qkv, float* __restrict__ ao) {
    const int b = blockIdx.z, h = blockIdx.y, qb = blockIdx.x;
    const int tid = threadIdx.x;
    const int lane = tid & 31, wave = tid >> 5;
    const int half = (lane >> 4) & 1, l16 = lane & 15;

    __shared__ unsigned short Qs[8][16 * 64];   // per-wave Q tiles (bf16)
    __shared__ unsigned short Ks[32 * 64];      // K tile row-major [key][d]
    __shared__ unsigned short VTs[64 * 32];     // V tile transposed [d][key]
    __shared__ unsigned short Ps[8][16 * 32];   // per-wave P tiles

    const size_t bstride = (size_t)2048 * 1536;
    const float* qb_p = qkv + b * bstride +        (size_t)h * 64;
    const float* kb_p = qkv + b * bstride + 512  + (size_t)h * 64;
    const float* vb_p = qkv + b * bstride + 1024 + (size_t)h * 64;

    const int q0 = qb * 128 + wave * 16;
    for (int idx = lane; idx < 16 * 64; idx += 32) {
        int r = idx >> 6, c = idx & 63;
        Qs[wave][idx] = f32_to_bf16(qb_p[(size_t)(q0 + r) * 1536 + c] * 0.125f);
    }

    v8f o[4] = {};
    float mrow[8], lrow[8];
    #pragma unroll
    for (int r = 0; r < 8; ++r) { mrow[r] = -1e30f; lrow[r] = 0.f; }

    for (int j0 = 0; j0 < 2048; j0 += 32) {
        __syncthreads();
        for (int idx = tid; idx < 32 * 64; idx += 256) {
            int r = idx >> 6, c = idx & 63;
            Ks[idx] = f32_to_bf16(kb_p[(size_t)(j0 + r) * 1536 + c]);
            VTs[c * 32 + r] = f32_to_bf16(vb_p[(size_t)(j0 + r) * 1536 + c]);
            if (j0 + 32 < 2048) __builtin_prefetch(kb_p + (size_t)(j0 + 32 + r) * 1536 + c, 0, 1);
        }
        __syncthreads();

        // scores: 16 queries x 32 keys, two 16x16 WMMA tiles, reduce d over 64
        v8f s0 = {}, s1 = {};
        #pragma unroll
        for (int kk = 0; kk < 64; kk += 32) {
            frag16 a, b0, b1;
            #pragma unroll
            for (int j = 0; j < 8; ++j) {
                int ka = 2 * (j & 3) + half * 8 + ((j >> 2) << 4) + kk;
                a.u[j]  = *(const unsigned int*)&Qs[wave][l16 * 64 + ka];
            int kb = 2 * (j & 3) + half * 8 + ((j >> 2) << 4) + kk;
                b0.u[j] = *(const unsigned int*)&Ks[l16 * 64 + kb];
                b1.u[j] = *(const unsigned int*)&Ks[(16 + l16) * 64 + kb];
            }
            s0 = __builtin_amdgcn_wmma_f32_16x16x32_bf16(false, a.v, false, b0.v,
                                                         (short)0, s0, false, false);
            s1 = __builtin_amdgcn_wmma_f32_16x16x32_bf16(false, a.v, false, b1.v,
                                                         (short)0, s1, false, false);
        }

        // online softmax; row = r + 8*half, cols striped across 16-lane group
        float p0[8], p1[8];
        #pragma unroll
        for (int r = 0; r < 8; ++r) {
            float tmax = fmaxf(s0[r], s1[r]);
            for (int off = 1; off < 16; off <<= 1)
                tmax = fmaxf(tmax, __shfl_xor(tmax, off, 16));
            float mnew  = fmaxf(mrow[r], tmax);
            float scale = __expf(mrow[r] - mnew);
            float e0 = __expf(s0[r] - mnew);
            float e1 = __expf(s1[r] - mnew);
            float rs = e0 + e1;
            for (int off = 1; off < 16; off <<= 1)
                rs += __shfl_xor(rs, off, 16);
            lrow[r] = lrow[r] * scale + rs;
            mrow[r] = mnew;
            o[0][r] *= scale; o[1][r] *= scale; o[2][r] *= scale; o[3][r] *= scale;
            p0[r] = e0; p1[r] = e1;
        }

        // C-layout -> A-fragment layout via per-wave LDS round trip
        #pragma unroll
        for (int r = 0; r < 8; ++r) {
            Ps[wave][(half * 8 + r) * 32 + l16]      = f32_to_bf16(p0[r]);
            Ps[wave][(half * 8 + r) * 32 + 16 + l16] = f32_to_bf16(p1[r]);
        }
        // same-wave DS ops are in-order; wait them out before fragment reads
        asm volatile("s_wait_dscnt 0" ::: "memory");

        frag16 pa;
        #pragma unroll
        for (int j = 0; j < 8; ++j) {
            int ka = 2 * (j & 3) + half * 8 + ((j >> 2) << 4);
            pa.u[j] = *(const unsigned int*)&Ps[wave][l16 * 32 + ka];
        }
        #pragma unroll
        for (int t = 0; t < 4; ++t) {
            frag16 vb;
            #pragma unroll
            for (int j = 0; j < 8; ++j) {
                int kb = 2 * (j & 3) + half * 8 + ((j >> 2) << 4);  // key index
                vb.u[j] = *(const unsigned int*)&VTs[(t * 16 + l16) * 32 + kb];
            }
            o[t] = __builtin_amdgcn_wmma_f32_16x16x32_bf16(false, pa.v, false, vb.v,
                                                           (short)0, o[t], false, false);
        }
    }

    float* obase = ao + b * (size_t)2048 * 512 + (size_t)h * 64;
    #pragma unroll
    for (int r = 0; r < 8; ++r) {
        float invl = 1.f / (lrow[r] + 1e-20f);
        int row = q0 + half * 8 + r;
        #pragma unroll
        for (int t = 0; t < 4; ++t)
            obase[(size_t)row * 512 + t * 16 + l16] = o[t][r] * invl;
    }
}

// ---------------------------------------------------------------------------
// Host-side pipeline. Workspace layout (136 MB):
//   Q: [0,48MB)  A: [48,64MB)  B: [64,80MB)  C: [80,96MB)
//   H: [96,128MB)  bf16 activation staging (up to 16M elems)
//   Wb:[128,136MB) bf16 weights (converted once per launch)
// ---------------------------------------------------------------------------
extern "C" void kernel_launch(void* const* d_in, const int* in_sizes, int n_in,
                              void* d_out, int out_size, void* d_ws, size_t ws_size,
                              hipStream_t stream) {
    const float* x        = (const float*)d_in[0];
    const float* x_prev   = (const float*)d_in[1];
    const float* mu       = (const float*)d_in[2];
    const float* in_w     = (const float*)d_in[3];
    const float* in_b     = (const float*)d_in[4];
    const float* out_w    = (const float*)d_in[5];
    const float* out_b    = (const float*)d_in[6];
    const float* ln1_g    = (const float*)d_in[7];
    const float* ln1_b    = (const float*)d_in[8];
    const float* time_mod = (const float*)d_in[9];
    const float* sm_g     = (const float*)d_in[10];
    const float* sm_b     = (const float*)d_in[11];
    const float* wr_w     = (const float*)d_in[12];
    const float* wr_b     = (const float*)d_in[13];
    const float* wv_w     = (const float*)d_in[14];
    const float* wv_b     = (const float*)d_in[15];
    const float* sc_g     = (const float*)d_in[16];
    const float* sc_b     = (const float*)d_in[17];
    const float* ln2_g    = (const float*)d_in[18];
    const float* ln2_b    = (const float*)d_in[19];
    const float* w1       = (const float*)d_in[20];
    const float* b1       = (const float*)d_in[21];
    const float* w2       = (const float*)d_in[22];
    const float* b2       = (const float*)d_in[23];
    const float* ln3_g    = (const float*)d_in[24];
    const float* ln3_b    = (const float*)d_in[25];
    float* out = (float*)d_out;

    char* ws = (char*)d_ws;
    float*          bufQ = (float*)(ws);                       // 48MB f32
    float*          bufA = (float*)(ws + (48u << 20));         // 16MB f32
    float*          bufB = (float*)(ws + (64u << 20));         // 16MB f32
    float*          bufC = (float*)(ws + (80u << 20));         // 16MB f32
    unsigned short* bufH = (unsigned short*)(ws + (96u << 20));  // 32MB bf16
    unsigned short* bufW = (unsigned short*)(ws + (128u << 20)); //  8MB bf16

    unsigned short* wb_in  = bufW;                 // 1536*512
    unsigned short* wb_out = wb_in  + 786432;      //  512*512
    unsigned short* wb_wr  = wb_out + 262144;      //  512*512
    unsigned short* wb_wv  = wb_wr  + 262144;      //  512*512
    unsigned short* wb_w1  = wb_wv  + 262144;      // 2048*512
    unsigned short* wb_w2  = wb_w1  + 1048576;     //  512*2048

    const int M = 4 * 2048;          // 8192 rows
    const int n = M * 512;           // 4,194,304 elements
    const int EB = 256;
    auto cvt = [&](const float* src, unsigned short* dst, int elems) {
        int n4 = elems / 4;
        cvt_bf16_kernel<<<(n4 + EB - 1) / EB, EB, 0, stream>>>(
            (const float4*)src, (uint2*)dst, n4);
    };

    // 0. weights -> bf16 (once per launch)
    cvt(in_w,  wb_in,  1536 * 512);
    cvt(out_w, wb_out,  512 * 512);
    cvt(wr_w,  wb_wr,   512 * 512);
    cvt(wv_w,  wb_wv,   512 * 512);
    cvt(w1,    wb_w1,  2048 * 512);
    cvt(w2,    wb_w2,   512 * 2048);

    // 1. shifted = mu*x + (1-mu)*x_prev  -> A
    shift_kernel<<<(n + EB - 1) / EB, EB, 0, stream>>>(x, x_prev, mu, bufA, n);
    // 2. qkv = shifted @ in_proj_w^T + b -> Q   [8192,1536]
    cvt(bufA, bufH, n);
    gemm_wmma_kernel<<<dim3(1536 / 64, M / 128), 256, 0, stream>>>(
        bufH, wb_in, in_b, bufQ, M, 1536, 512, 0);
    // 3. attention -> B (raw, pre-projection)
    flash_attn_kernel<<<dim3(16, 8, 4), 256, 0, stream>>>(bufQ, bufB);
    // 4. out projection -> C
    cvt(bufB, bufH, n);
    gemm_wmma_kernel<<<dim3(512 / 64, M / 128), 256, 0, stream>>>(
        bufH, wb_out, out_b, bufC, M, 512, 512, 0);
    // 5. x1 = LN(shifted + proj) -> B
    ln_kernel<<<M, 256, 0, stream>>>(bufA, bufC, ln1_g, ln1_b, bufB);
    // 6. sequential scan -> Q (first 16MB)
    scan_kernel<<<4, 512, 0, stream>>>(bufB, time_mod, bufQ);
    // 7. merged = LN(outs) -> A
    ln_kernel<<<M, 256, 0, stream>>>(bufQ, nullptr, sm_g, sm_b, bufA);
    // 8. r = merged @ wr^T -> C ;  v = merged @ wv^T -> Q (one shared cvt)
    cvt(bufA, bufH, n);
    gemm_wmma_kernel<<<dim3(512 / 64, M / 128), 256, 0, stream>>>(
        bufH, wb_wr, wr_b, bufC, M, 512, 512, 0);
    gemm_wmma_kernel<<<dim3(512 / 64, M / 128), 256, 0, stream>>>(
        bufH, wb_wv, wv_b, bufQ, M, 512, 512, 0);
    // 9. coupled = LN(sigmoid(r)*v) -> C (both steps in place)
    sigmul_kernel<<<(n + EB - 1) / EB, EB, 0, stream>>>(bufC, bufQ, bufC, n);
    ln_kernel<<<M, 256, 0, stream>>>(bufC, nullptr, sc_g, sc_b, bufC);
    // 10. x2 = LN(x1 + coupled) -> B (in place over x1)
    ln_kernel<<<M, 256, 0, stream>>>(bufB, bufC, ln2_g, ln2_b, bufB);
    // 11. hidden = relu(x2 @ w1^T + b1) -> Q (spans 64MB: Q+A, both dead)
    cvt(bufB, bufH, n);
    gemm_wmma_kernel<<<dim3(2048 / 64, M / 128), 256, 0, stream>>>(
        bufH, wb_w1, b1, bufQ, M, 2048, 512, 1);
    // 12. ffn = hidden @ w2^T + b2 -> C
    cvt(bufQ, bufH, 4 * n);                      // 16M elements -> 32MB bf16
    gemm_wmma_kernel<<<dim3(512 / 64, M / 128), 256, 0, stream>>>(
        bufH, wb_w2, b2, bufC, M, 512, 2048, 0);
    // 13. out = LN(x2 + ffn) -> d_out
    ln_kernel<<<M, 256, 0, stream>>>(bufB, bufC, ln3_g, ln3_b, out);
}
